// AutoRegressiveDistribution_7808250544657
// MI455X (gfx1250) — compile-verified, matrix operand order checked by reading, not run on hardware
//
#include <hip/hip_runtime.h>
#include <hip/hip_bf16.h>
#include <math.h>

#define D_LAT 64
#define HID   1024
#define CTXN  256
#define BATCH 1024
#define NS    4
#define ROWS  (NS*BATCH)            /* 4096 */
#define ROWS_PER_BLK 16
#define NBLK  (ROWS/ROWS_PER_BLK)   /* 256 */
#define NTHR  256
#define HSTRIDE 1025                /* 1024 + 1 pad: kills 16-way LDS bank conflicts */

typedef float  v2f  __attribute__((ext_vector_type(2)));
typedef float  v8f  __attribute__((ext_vector_type(8)));
typedef __bf16 v16bf __attribute__((ext_vector_type(16)));

#if __has_builtin(__builtin_amdgcn_wmma_f32_16x16x4_f32)
#define HAVE_WMMA_F32 1
#else
#define HAVE_WMMA_F32 0
#endif

__global__ __launch_bounds__(NTHR, 1)
void made_ar_kernel(const float* __restrict__ context,
                    const float* __restrict__ eps,
                    const float* __restrict__ W1,
                    const float* __restrict__ b1,
                    const float* __restrict__ Wc,
                    const float* __restrict__ Wout,
                    const float* __restrict__ bout,
                    float* __restrict__ out)
{
    __shared__ float hpre[ROWS_PER_BLK * HSTRIDE];  /* ~64 KB, lives in LDS all 64 steps */
    __shared__ float red_mu[NTHR];
    __shared__ float red_ps[NTHR];
    __shared__ float zsh[ROWS_PER_BLK];

    const int tid    = threadIdx.x;
    const int lane   = tid & 31;
    const int wave   = tid >> 5;      /* 0..7 */
    const int laneM  = lane & 15;     /* M (A/C) or N (B/C) index */
    const int laneHi = lane >> 4;     /* 0/1: K-half select */

    const int r0 = blockIdx.x * ROWS_PER_BLK;   /* 16 consecutive flat rows (same s) */
    const int b0 = r0 % BATCH;                  /* contiguous context rows */

    /* ---------------- Phase 1: hpre = context @ Wc^T + b1 via WMMA ----------------
       A = context rows b0..b0+15 (16 x 256), B = Wc^T (256 x 1024).
       Each wave owns a 128-column slab: 8 N-tiles x (K=256 / step) WMMAs.           */
    const float* arow = context + (size_t)(b0 + laneM) * CTXN;
    for (int nt = 0; nt < 8; ++nt) {
        const int n = wave * 128 + nt * 16 + laneM;
        const float* brow = Wc + (size_t)n * CTXN;
        v8f acc = {};
#if HAVE_WMMA_F32
        for (int kt = 0; kt < 64; ++kt) {
            const int k = kt * 4 + laneHi * 2;          /* 16x4 f32 A layout */
            v2f av; av.x = arow[k]; av.y = arow[k + 1];
            v2f bv; bv.x = brow[k]; bv.y = brow[k + 1];
            acc = __builtin_amdgcn_wmma_f32_16x16x4_f32(
                      false, av, false, bv, (short)0, acc, false, false);
        }
#else
        for (int kt = 0; kt < 8; ++kt) {
            const int k0 = kt * 32;                     /* 16x32 bf16 A layout */
            v16bf av, bv;
#pragma unroll
            for (int j = 0; j < 8; ++j) {
                av[j]     = (__bf16)arow[k0 + laneHi * 8 + j];
                av[8 + j] = (__bf16)arow[k0 + 16 + laneHi * 8 + j];
                bv[j]     = (__bf16)brow[k0 + laneHi * 8 + j];
                bv[8 + j] = (__bf16)brow[k0 + 16 + laneHi * 8 + j];
            }
            acc = __builtin_amdgcn_wmma_f32_16x16x32_bf16(
                      false, av, false, bv, (short)0, acc, false, false);
        }
#endif
        const float bias = b1[n];
#pragma unroll
        for (int v = 0; v < 8; ++v) {
            const int m = v + laneHi * 8;               /* C tile: vgpr v -> row m */
            hpre[m * HSTRIDE + n] = acc[v] + bias;
        }
    }
    __syncthreads();

    /* ---------------- Phase 2: 64-step autoregressive chain ----------------
       thread = (row, seg): row = tid&15 of the 16 rows, seg = tid>>4 owns a
       64-wide slice of the 1024 hidden units.                                   */
    const int row = tid & 15;
    const int seg = tid >> 4;
    const int h0  = seg * 64;
    const int mh0 = h0 % 63 + 1;   /* mh(h) = h % 63 + 1, tracked incrementally */

    for (int i = 0; i < D_LAT; ++i) {
        /* masked dual dot-product: mu (Wout row i) and prescale (Wout row 64+i);
           both share the output mask  mh(h) <= i.  relu applied on the fly.     */
        const float* wmu = Wout + (size_t)i * HID;
        const float* wps = Wout + (size_t)(D_LAT + i) * HID;
        float pm = 0.f, pp = 0.f;
        int mh = mh0;
#pragma unroll 8
        for (int j = 0; j < 64; ++j) {
            const int h = h0 + j;
            float r = hpre[row * HSTRIDE + h];
            r = r > 0.f ? r : 0.f;
            const float msk = (mh <= i) ? 1.f : 0.f;
            const float rm = r * msk;
            pm = fmaf(rm, wmu[h], pm);
            pp = fmaf(rm, wps[h], pp);
            if (++mh == 64) mh = 1;
        }
        red_mu[tid] = pm;
        red_ps[tid] = pp;
        __syncthreads();

        if (tid < ROWS_PER_BLK) {       /* one thread per row finishes the step */
            float sm = bout[i];
            float sp = bout[D_LAT + i];
#pragma unroll
            for (int sgi = 0; sgi < 16; ++sgi) {
                sm += red_mu[sgi * 16 + tid];
                sp += red_ps[sgi * 16 + tid];
            }
            /* stable softplus: max(x,0) + log1p(exp(-|x|)) */
            const float scale = fmaxf(sp, 0.f) + log1pf(expf(-fabsf(sp)));
            const int    r = r0 + tid;
            const size_t o = (size_t)r * D_LAT + i;
            const float  z = fmaf(scale, eps[o], sm);
            zsh[tid] = z;
            out[o]                          = z;      /* z_sample  */
            out[(size_t)ROWS * D_LAT + o]   = sm;     /* mu        */
            out[(size_t)2 * ROWS * D_LAT + o] = scale;/* scale     */
        }
        __syncthreads();

        /* rank-1 update: hpre[:,h] += z_i * W1[h][i] * (mh(h) >= i+1) */
        const float z = zsh[row];
        int mh2 = mh0;
#pragma unroll 8
        for (int j = 0; j < 64; ++j) {
            const int h = h0 + j;
            const float w = (mh2 >= i + 1) ? W1[(size_t)h * D_LAT + i] : 0.f;
            hpre[row * HSTRIDE + h] = fmaf(z, w, hpre[row * HSTRIDE + h]);
            if (++mh2 == 64) mh2 = 1;
        }
        __syncthreads();
    }
}

extern "C" void kernel_launch(void* const* d_in, const int* in_sizes, int n_in,
                              void* d_out, int out_size, void* d_ws, size_t ws_size,
                              hipStream_t stream) {
    (void)in_sizes; (void)n_in; (void)out_size; (void)d_ws; (void)ws_size;
    const float* context = (const float*)d_in[0];
    const float* eps     = (const float*)d_in[1];
    const float* W1      = (const float*)d_in[2];
    const float* b1      = (const float*)d_in[3];
    const float* Wc      = (const float*)d_in[4];
    const float* Wout    = (const float*)d_in[5];
    const float* bout    = (const float*)d_in[6];
    made_ar_kernel<<<NBLK, NTHR, 0, stream>>>(context, eps, W1, b1, Wc, Wout, bout,
                                              (float*)d_out);
}
